// MoE_58162447122836
// MI455X (gfx1250) — matrix operand order repaired and checked by reading, not compiled, not run on hardware
//
#include <hip/hip_runtime.h>
#include <math.h>

// ---------------------------------------------------------------------------
// Types for CDNA5 WMMA (wave32): v_wmma_f32_16x16x32_bf16
// ---------------------------------------------------------------------------
typedef __attribute__((ext_vector_type(16))) __bf16 v16bf;
typedef __attribute__((ext_vector_type(8)))  __bf16 v8bf;
typedef __attribute__((ext_vector_type(8)))  float  v8f;

union V16U { v16bf v; v8bf h[2]; };

// Problem constants
#define TN   12544   // B*S tokens
#define DN   768
#define HN   128
#define EN   16
#define BN   64
#define KFC4 25088   // S*H
#define OUTN 18
#define SPLITK 49    // 784 k-tiles / 16

// LDS row strides, padded so ds_load_b128 A-fragments are bank-conflict free:
// stride(dwords) % 64 must be != 0; 776*2/4 = 388 = 6*64+4, 136*2/4 = 68 = 64+4.
#define XSP  776     // padded x-tile row stride (elements)
#define HBS  136     // padded activation-tile row stride (elements)

__device__ __forceinline__ unsigned short f2bf(float f) {
    unsigned int u = __float_as_uint(f);
    u = (u + 0x7FFFu + ((u >> 16) & 1u)) >> 16;   // round-to-nearest-even
    return (unsigned short)u;
}

// Load a 16-element bf16 fragment from two 16-byte-aligned halves.
__device__ __forceinline__ v16bf load16(const unsigned short* p0,
                                        const unsigned short* p1) {
    V16U u;
    u.h[0] = *(const v8bf*)p0;
    u.h[1] = *(const v8bf*)p1;
    return u.v;
}

__device__ __forceinline__ v16bf loadB(const unsigned short* p) {
    union { v16bf v; unsigned short u[16]; } t;
#pragma unroll
    for (int s = 0; s < 16; ++s) t.u[s] = p[(size_t)(s < 8 ? s : s + 8) * HN];
    return t.v;
}

__device__ __forceinline__ v8f wmma_bf16(v16bf a, v16bf b, v8f c) {
    return __builtin_amdgcn_wmma_f32_16x16x32_bf16(
        /*neg_a=*/false, a, /*neg_b=*/false, b,
        /*c_mod=*/(short)0, c, /*reuse_a=*/false, /*reuse_b=*/false);
}

// ---------------------------------------------------------------------------
// Kernel: elementwise fp32 -> bf16
// ---------------------------------------------------------------------------
__global__ void cvt_bf16_kernel(const float* __restrict__ s,
                                unsigned short* __restrict__ d, int n) {
    int i = blockIdx.x * blockDim.x + threadIdx.x;
    int stride = gridDim.x * blockDim.x;
    for (; i < n; i += stride) d[i] = f2bf(s[i]);
}

// ---------------------------------------------------------------------------
// Kernel: gating -> comb[t][e]  (softmax, top-2 indices, slot weights = probs[:2])
// ---------------------------------------------------------------------------
__global__ void gate_kernel(const float* __restrict__ x,
                            const float* __restrict__ gw,
                            const float* __restrict__ gb,
                            float* __restrict__ comb) {
    int t = blockIdx.x * blockDim.x + threadIdx.x;
    if (t >= TN) return;
    float l[EN];
#pragma unroll
    for (int e = 0; e < EN; ++e) l[e] = gb[e];
    const float* xr = x + (size_t)t * DN;
    for (int d = 0; d < DN; ++d) {
        float xv = xr[d];
        const float* wr = gw + d * EN;
#pragma unroll
        for (int e = 0; e < EN; ++e) l[e] += xv * wr[e];
    }
    float mx = l[0];
#pragma unroll
    for (int e = 1; e < EN; ++e) mx = l[e] > mx ? l[e] : mx;
    float Z = 0.f;
#pragma unroll
    for (int e = 0; e < EN; ++e) Z += expf(l[e] - mx);
    float p0 = expf(l[0] - mx) / Z;      // slot_w = probs[..., :2]
    float p1 = expf(l[1] - mx) / Z;
    int i0 = 0; float b0 = l[0];
#pragma unroll
    for (int e = 1; e < EN; ++e) if (l[e] > b0) { b0 = l[e]; i0 = e; }
    int i1 = -1; float b1v = -3.4e38f;
#pragma unroll
    for (int e = 0; e < EN; ++e) if (e != i0 && l[e] > b1v) { b1v = l[e]; i1 = e; }
#pragma unroll
    for (int e = 0; e < EN; ++e)
        comb[(size_t)t * EN + e] = (e == i0 ? p0 : 0.f) + (e == i1 ? p1 : 0.f);
}

// ---------------------------------------------------------------------------
// Fused MoE + fc2 + fc3 kernel.
// Grid: 784 workgroups x 256 threads (8 waves). Each WG owns 16 tokens.
// Wave w computes output columns [16w, 16w+16).
// The 16x768 x-tile is staged once per WG into LDS with CDNA5 async
// global->LDS copies (ASYNCcnt), then reused by all 8 waves for all experts.
// ---------------------------------------------------------------------------
__global__ void __launch_bounds__(256)
moe_fused_kernel(const unsigned short* __restrict__ xbf,
                 const unsigned short* __restrict__ w1bf, const float* __restrict__ b1,
                 const unsigned short* __restrict__ w2bf, const float* __restrict__ b2,
                 const unsigned short* __restrict__ w3bf, const float* __restrict__ b3,
                 const unsigned short* __restrict__ fc2bf, const float* __restrict__ fc2b,
                 const unsigned short* __restrict__ fc3bf, const float* __restrict__ fc3b,
                 const float* __restrict__ comb,
                 unsigned short* __restrict__ y3bf) {
    __shared__ unsigned short xs[16 * XSP];     // staged 16x768 x-tile (padded rows)
    __shared__ unsigned short hb[2][16 * HBS];  // 16x128 bf16 activation tiles
    __shared__ float combT[16 * EN];
    __shared__ int eflag[EN];

    const int rt   = blockIdx.x;        // token tile 0..783
    const int tid  = threadIdx.x;
    const int lane = tid & 31;
    const int wv   = tid >> 5;          // 0..7 -> n-tile
    const int lm   = lane & 15;
    const int hi   = lane >> 4;
    const int ng   = wv * 16 + lm;      // global output column 0..127
    const int row0 = rt * 16;

    // ---- async-stage the x-tile: 16 rows x 1536B, 6 x b128 per thread ----
    {
        unsigned xs_base = (unsigned)(unsigned long long)(const void*)&xs[0];
        unsigned long long gbase =
            (unsigned long long)(const void*)(xbf + (size_t)row0 * DN);
#pragma unroll
        for (int i = 0; i < 6; ++i) {
            int g   = i * 256 + tid;        // b128 chunk id, 0..1535
            int row = g / 96;               // 96 chunks of 16B per 1536B row
            int c   = g - row * 96;
            unsigned           loff = xs_base + (unsigned)(row * (XSP * 2) + c * 16);
            unsigned long long ga   = gbase + (unsigned long long)(row * 1536 + c * 16);
            asm volatile("global_load_async_to_lds_b128 %0, %1, off"
                         :: "v"(loff), "v"(ga) : "memory");
        }
    }

    combT[tid & 255] = comb[(size_t)(row0 + (tid >> 4)) * EN + (tid & 15)];
    __syncthreads();
    if (tid < EN) {
        int f = 0;
#pragma unroll
        for (int m = 0; m < 16; ++m) f |= (combT[m * EN + tid] != 0.f);
        eflag[tid] = f;
    }
    asm volatile("s_wait_asynccnt 0x0" ::: "memory");
    __syncthreads();

    float moe[8] = {0.f, 0.f, 0.f, 0.f, 0.f, 0.f, 0.f, 0.f};

    for (int e = 0; e < EN; ++e) {
        if (!eflag[e]) continue;        // block-uniform: exact-math expert skip

        // ---- layer 1: X(16x768) @ W1[e](768x128), A from LDS ----
        v8f acc = {0.f, 0.f, 0.f, 0.f, 0.f, 0.f, 0.f, 0.f};
        const unsigned short* w1e = w1bf + (size_t)e * DN * HN;
#pragma unroll 4
        for (int kt = 0; kt < DN / 32; ++kt) {
            const unsigned short* pa = &xs[lm * XSP + kt * 32 + 8 * hi];
            v16bf a = load16(pa, pa + 16);
            int kb = kt * 32 + 8 * hi;
            const unsigned short* pb = w1e + (size_t)kb * HN + wv * 16 + lm;
            v16bf b = loadB(pb);
            acc = wmma_bf16(a, b, acc);
        }
        {
            float bb = b1[e * HN + ng];
#pragma unroll
            for (int v = 0; v < 8; ++v) {
                float t = acc[v] + bb; t = t > 0.f ? t : 0.f;
                hb[0][(v + 8 * hi) * HBS + ng] = f2bf(t);
            }
        }
        __syncthreads();

        // ---- layer 2: h1(16x128) @ W2[e](128x128) ----
        v8f acc2 = {0.f, 0.f, 0.f, 0.f, 0.f, 0.f, 0.f, 0.f};
        const unsigned short* w2e = w2bf + (size_t)e * HN * HN;
#pragma unroll
        for (int kt = 0; kt < HN / 32; ++kt) {
            const unsigned short* pa = &hb[0][lm * HBS + kt * 32 + 8 * hi];
            v16bf a = load16(pa, pa + 16);
            int kb = kt * 32 + 8 * hi;
            const unsigned short* pb = w2e + (size_t)kb * HN + wv * 16 + lm;
            v16bf b = loadB(pb);
            acc2 = wmma_bf16(a, b, acc2);
        }
        {
            float bb = b2[e * HN + ng];
#pragma unroll
            for (int v = 0; v < 8; ++v) {
                float t = acc2[v] + bb; t = t > 0.f ? t : 0.f;
                hb[1][(v + 8 * hi) * HBS + ng] = f2bf(t);
            }
        }
        __syncthreads();

        // ---- layer 3: h2(16x128) @ W3[e](128x128), weighted accumulate ----
        v8f acc3 = {0.f, 0.f, 0.f, 0.f, 0.f, 0.f, 0.f, 0.f};
        const unsigned short* w3e = w3bf + (size_t)e * HN * HN;
#pragma unroll
        for (int kt = 0; kt < HN / 32; ++kt) {
            const unsigned short* pa = &hb[1][lm * HBS + kt * 32 + 8 * hi];
            v16bf a = load16(pa, pa + 16);
            int kb = kt * 32 + 8 * hi;
            const unsigned short* pb = w3e + (size_t)kb * HN + wv * 16 + lm;
            v16bf b = loadB(pb);
            acc3 = wmma_bf16(a, b, acc3);
        }
        {
            float bb = b3[e * HN + ng];
#pragma unroll
            for (int v = 0; v < 8; ++v) {
                float t = acc3[v] + bb; t = t > 0.f ? t : 0.f;
                moe[v] += combT[(v + 8 * hi) * EN + e] * t;
            }
        }
        // next expert's layer-1 write targets hb[0] (last read pre-barrier in
        // layer 2); barrier after that write protects hb[1] for layer 2.
    }

    // ---- y = relu(moe) -> fc2 ----
#pragma unroll
    for (int v = 0; v < 8; ++v) {
        float t = moe[v] > 0.f ? moe[v] : 0.f;
        hb[0][(v + 8 * hi) * HBS + ng] = f2bf(t);
    }
    __syncthreads();

    v8f acc4 = {0.f, 0.f, 0.f, 0.f, 0.f, 0.f, 0.f, 0.f};
#pragma unroll
    for (int kt = 0; kt < HN / 32; ++kt) {
        const unsigned short* pa = &hb[0][lm * HBS + kt * 32 + 8 * hi];
        v16bf a = load16(pa, pa + 16);
        int kb = kt * 32 + 8 * hi;
        const unsigned short* pb = fc2bf + (size_t)kb * HN + wv * 16 + lm;
        v16bf b = loadB(pb);
        acc4 = wmma_bf16(a, b, acc4);
    }
    {
        float bb = fc2b[ng];
#pragma unroll
        for (int v = 0; v < 8; ++v) {
            float t = acc4[v] + bb; t = t > 0.f ? t : 0.f;
            hb[1][(v + 8 * hi) * HBS + ng] = f2bf(t);
        }
    }
    __syncthreads();

    // ---- fc3, emit y3 (bf16, row-major (12544,128) == flat (64,25088)) ----
    v8f acc5 = {0.f, 0.f, 0.f, 0.f, 0.f, 0.f, 0.f, 0.f};
#pragma unroll
    for (int kt = 0; kt < HN / 32; ++kt) {
        const unsigned short* pa = &hb[1][lm * HBS + kt * 32 + 8 * hi];
        v16bf a = load16(pa, pa + 16);
        int kb = kt * 32 + 8 * hi;
        const unsigned short* pb = fc3bf + (size_t)kb * HN + wv * 16 + lm;
        v16bf b = loadB(pb);
        acc5 = wmma_bf16(a, b, acc5);
    }
    {
        float bb = fc3b[ng];
#pragma unroll
        for (int v = 0; v < 8; ++v) {
            float t = acc5[v] + bb; t = t > 0.f ? t : 0.f;
            y3bf[(size_t)(row0 + v + 8 * hi) * HN + ng] = f2bf(t);
        }
    }
}

// ---------------------------------------------------------------------------
// fc4 split-K WMMA: (64 x 25088) @ (25088 x 128), partials per K-split.
// Grid: 49 splits * 4 row-tiles * 8 n-tiles, 32 threads (1 wave) each.
// ---------------------------------------------------------------------------
__global__ void __launch_bounds__(32)
fc4_partial_kernel(const unsigned short* __restrict__ y3bf,
                   const unsigned short* __restrict__ fc4bf,
                   float* __restrict__ part) {
    int blk = blockIdx.x;
    int sk  = blk >> 5;       // 0..48
    int rem = blk & 31;
    int rtb = rem >> 3;       // 0..3 (rows 16*rtb .. +15)
    int nt  = rem & 7;        // 0..7
    int lane = threadIdx.x;
    int lm = lane & 15, hi = lane >> 4;

    v8f acc = {0.f, 0.f, 0.f, 0.f, 0.f, 0.f, 0.f, 0.f};
    const unsigned short* arow = y3bf + (size_t)(rtb * 16 + lm) * KFC4;
#pragma unroll 4
    for (int kk = 0; kk < 16; ++kk) {
        int kt = sk * 16 + kk;                       // k-tile 0..783
        const unsigned short* pa = arow + kt * 32 + 8 * hi;
        v16bf a = load16(pa, pa + 16);
        int kb = kt * 32 + 8 * hi;
        const unsigned short* pb = fc4bf + (size_t)kb * HN + nt * 16 + lm;
        v16bf b = loadB(pb);
        acc = wmma_bf16(a, b, acc);
    }
    float* dst = part + (size_t)sk * (BN * HN);
#pragma unroll
    for (int v = 0; v < 8; ++v)
        dst[(rtb * 16 + v + 8 * hi) * HN + nt * 16 + lm] = acc[v];
}

// ---------------------------------------------------------------------------
// Head: reduce fc4 partials + bias + relu, then fc5 (relu) and fc6 on VALU.
// Grid: 64 blocks x 128 threads.
// ---------------------------------------------------------------------------
__global__ void __launch_bounds__(128)
head_kernel(const float* __restrict__ part, const float* __restrict__ fc4b,
            const float* __restrict__ fc5w, const float* __restrict__ fc5b,
            const float* __restrict__ fc6w, const float* __restrict__ fc6b,
            float* __restrict__ out) {
    __shared__ float y4[HN], y5[HN];
    int r = blockIdx.x, t = threadIdx.x;
    float s = fc4b[t];
    for (int sk = 0; sk < SPLITK; ++sk) s += part[(size_t)sk * (BN * HN) + r * HN + t];
    y4[t] = s > 0.f ? s : 0.f;
    __syncthreads();
    float a = fc5b[t];
#pragma unroll 8
    for (int k = 0; k < HN; ++k) a += y4[k] * fc5w[k * HN + t];
    y5[t] = a > 0.f ? a : 0.f;
    __syncthreads();
    if (t < OUTN) {
        float o = fc6b[t];
#pragma unroll 8
        for (int k = 0; k < HN; ++k) o += y5[k] * fc6w[k * OUTN + t];
        out[r * OUTN + t] = o;
    }
}

// ---------------------------------------------------------------------------
extern "C" void kernel_launch(void* const* d_in, const int* in_sizes, int n_in,
                              void* d_out, int out_size, void* d_ws, size_t ws_size,
                              hipStream_t stream) {
    (void)in_sizes; (void)n_in; (void)out_size; (void)ws_size;
    const float* x      = (const float*)d_in[0];
    const float* gate_w = (const float*)d_in[1];
    const float* gate_b = (const float*)d_in[2];
    const float* w1     = (const float*)d_in[3];
    const float* b1     = (const float*)d_in[4];
    const float* w2     = (const float*)d_in[5];
    const float* b2     = (const float*)d_in[6];
    const float* w3     = (const float*)d_in[7];
    const float* b3     = (const float*)d_in[8];
    const float* fc2w   = (const float*)d_in[9];
    const float* fc2b   = (const float*)d_in[10];
    const float* fc3w   = (const float*)d_in[11];
    const float* fc3b   = (const float*)d_in[12];
    const float* fc4w   = (const float*)d_in[13];
    const float* fc4b   = (const float*)d_in[14];
    const float* fc5w   = (const float*)d_in[15];
    const float* fc5b   = (const float*)d_in[16];
    const float* fc6w   = (const float*)d_in[17];
    const float* fc6b   = (const float*)d_in[18];

    char* ws = (char*)d_ws;
    size_t off = 0;
    auto alloc = [&](size_t bytes) -> void* {
        void* p = ws + off;
        off = (off + bytes + 255) & ~(size_t)255;
        return p;
    };
    unsigned short* xbf   = (unsigned short*)alloc((size_t)TN * DN * 2);
    unsigned short* w1bf  = (unsigned short*)alloc((size_t)EN * DN * HN * 2);
    unsigned short* w2bf  = (unsigned short*)alloc((size_t)EN * HN * HN * 2);
    unsigned short* w3bf  = (unsigned short*)alloc((size_t)EN * HN * HN * 2);
    unsigned short* fc2bf = (unsigned short*)alloc((size_t)HN * HN * 2);
    unsigned short* fc3bf = (unsigned short*)alloc((size_t)HN * HN * 2);
    unsigned short* fc4bf = (unsigned short*)alloc((size_t)KFC4 * HN * 2);
    float*          comb  = (float*)alloc((size_t)TN * EN * 4);
    unsigned short* y3bf  = (unsigned short*)alloc((size_t)TN * HN * 2);
    float*          part  = (float*)alloc((size_t)SPLITK * BN * HN * 4);

    auto cvt = [&](const float* s, unsigned short* d, int n) {
        int blocks = (n + 2047) / 2048;
        cvt_bf16_kernel<<<blocks, 256, 0, stream>>>(s, d, n);
    };
    cvt(x,    xbf,   TN * DN);
    cvt(w1,   w1bf,  EN * DN * HN);
    cvt(w2,   w2bf,  EN * HN * HN);
    cvt(w3,   w3bf,  EN * HN * HN);
    cvt(fc2w, fc2bf, HN * HN);
    cvt(fc3w, fc3bf, HN * HN);
    cvt(fc4w, fc4bf, KFC4 * HN);

    gate_kernel<<<(TN + 255) / 256, 256, 0, stream>>>(x, gate_w, gate_b, comb);

    moe_fused_kernel<<<TN / 16, 256, 0, stream>>>(
        xbf, w1bf, b1, w2bf, b2, w3bf, b3, fc2bf, fc2b, fc3bf, fc3b, comb, y3bf);

    fc4_partial_kernel<<<SPLITK * 32, 32, 0, stream>>>(y3bf, fc4bf, part);

    head_kernel<<<BN, 128, 0, stream>>>(part, fc4b, fc5w, fc5b, fc6w, fc6b,
                                        (float*)d_out);
}
